// point_cloud_decoder_40750649704492
// MI455X (gfx1250) — matrix operand order repaired and checked by reading, not compiled, not run on hardware
//
#include <hip/hip_runtime.h>
#include <hip/hip_bf16.h>
#include <math.h>

// ---------------------------------------------------------------------------
// Problem constants (from reference)
// ---------------------------------------------------------------------------
constexpr int GG   = 16;    // graphs
constexpr int MM   = 512;   // nodes per graph
constexpr int KK   = 32;    // neighbors per node
constexpr int DD   = 128;   // feature dim (= DIN)
constexpr int HH   = 4;     // heads
constexpr int DH   = 32;    // dim per head
constexpr int RR   = 50;    // radial basis size
constexpr int RPAD = 64;    // padded K-dim for the edge GEMM
constexpr int NGP  = 27;    // grid points
constexpr int LL   = 2;     // layers
constexpr int OUTN = 100;
constexpr int NODES = GG * MM;          // 8192
constexpr int EDGES = NODES * KK;       // 262144
constexpr float CUT   = 2.0f;
constexpr float RSTEP = CUT / (RR - 1);             // linspace step
constexpr float RCOEF = -0.5f / (RSTEP * RSTEP);    // gaussian coeff
constexpr float SCALE = 0.17677669529663687f;       // 1/sqrt(32)

typedef __attribute__((ext_vector_type(16))) _Float16 v16h;
typedef __attribute__((ext_vector_type(8)))  float    v8f;

#define DEV __device__ __forceinline__

DEV float gelu_f(float v) {            // exact erf GELU (torch default)
    return 0.5f * v * (1.0f + erff(v * 0.70710678118654752f));
}

// A-fragment loader (16-bit A 16x32, this file's contiguous k order, matched by B): lane owns one row; halves
// h[2v+s] = A[row][koff + 2*v + s], koff = (lane>=16)*16.
// Ar must point at A[row] + koff; k0 passed per step. Compiles to b128 loads
// + v_cvt_pk_f16_f32 (verified in round-1 asm).
DEV v16h load_a_frag(const float* __restrict__ Ar, int k0) {
    v16h av;
#pragma unroll
    for (int v = 0; v < 8; ++v) {
        const int kk = k0 + (v << 1);
        av[2 * v]     = (_Float16)Ar[kk];
        av[2 * v + 1] = (_Float16)Ar[kk + 1];
    }
    return av;
}

// ---------------------------------------------------------------------------
// Generic WMMA GEMM: C[M,N] = act(A[M,128] @ B[128,N] + bias)
// K hardcoded to 128 (true for every use) -> fully unrolled 4 x wmma.
// 256 threads = 8 waves; block tile 32 rows x 64 cols; one 16x16 tile / wave.
// B panel staged ONCE per block into LDS, transposed to [col][k] halves so a
// wave's B fragment is one contiguous 32B LDS read. N-guard is a branch-free
// clamped column at staging; stores guarded.
// ---------------------------------------------------------------------------
__global__ void wmma_gemm_kernel(const float* __restrict__ A,
                                 const float* __restrict__ B,
                                 const float* __restrict__ bias,
                                 float* __restrict__ C,
                                 int N, int ldb, int ldc, int act) {
    __shared__ alignas(32) _Float16 lB[64 * 128];   // [col-in-block][k], 16KB

    const int wid  = threadIdx.x >> 5;
    const int lane = threadIdx.x & 31;
    const int row0 = blockIdx.x * 32 + (wid >> 2) * 16;
    const int col0 = blockIdx.y * 64;

    // ---- stage B panel (128 k x 64 cols), transposed, f32 -> f16 ----
    for (int i = threadIdx.x; i < 128 * 16; i += 256) {
        const int k  = i >> 4;
        const int c4 = (i & 15) << 2;
#pragma unroll
        for (int j = 0; j < 4; ++j) {
            const int col = col0 + c4 + j;
            const int cc  = col < N ? col : (N - 1);       // branch-free clamp
            lB[(c4 + j) * 128 + k] = (_Float16)B[(long)k * ldb + cc];
        }
    }
    __syncthreads();

    const int arow  = row0 + (lane & 15);
    const int koff  = (lane >> 4) << 4;
    const int bcolL = ((wid & 3) << 4) + (lane & 15);      // 0..63 in block
    const int bkoff = (lane >> 4) << 4;
    const float* Arow = A + (long)arow * 128 + koff;

    v8f acc = {};
#pragma unroll
    for (int k0 = 0; k0 < 128; k0 += 32) {
        const v16h av = load_a_frag(Arow, k0);
        const v16h bv = *(const v16h*)&lB[bcolL * 128 + k0 + bkoff];
        acc = __builtin_amdgcn_wmma_f32_16x16x32_f16(
            false, av, false, bv, (short)0, acc, false, false);
    }

    const int bcol = col0 + bcolL;
    if (bcol < N) {
        const int crow0 = row0 + ((lane >> 4) << 3);
        const float bb = bias ? bias[bcol] : 0.0f;
#pragma unroll
        for (int r = 0; r < 8; ++r) {
            float v = acc[r] + bb;
            if (act) v = gelu_f(v);
            C[(long)(crow0 + r) * ldc + bcol] = v;
        }
    }
}

// ---------------------------------------------------------------------------
// Edge-embedding WMMA GEMM: e[E,128] = RBF(dist)[E,64pad] @ Weff[64,128] + be
// A tiles synthesized in registers: A(row,r) = exp(RCOEF*(dist[row]-r*RSTEP)^2)
// for r<50, else 0 (Weff rows 50..63 are zero-padded). Each lane owns one edge
// row -> one dist load per tile. Weff panel staged transposed in LDS once.
// ---------------------------------------------------------------------------
DEV float rbf_val(float d, int r) {
    if (r >= RR) return 0.0f;
    const float t = d - (float)r * RSTEP;
    return expf(RCOEF * t * t);
}

__global__ void edge_wmma_gemm_kernel(const float* __restrict__ dist,
                                      const float* __restrict__ Weff,
                                      const float* __restrict__ be,
                                      float* __restrict__ E) {
    __shared__ alignas(32) _Float16 lB[64 * 64];    // [col-in-block][k], 8KB

    const int wid  = threadIdx.x >> 5;
    const int lane = threadIdx.x & 31;
    const int row0 = blockIdx.x * 32 + (wid >> 2) * 16;
    const int col0 = blockIdx.y * 64;

    for (int i = threadIdx.x; i < 64 * 16; i += 256) {
        const int k  = i >> 4;
        const int c4 = (i & 15) << 2;
#pragma unroll
        for (int j = 0; j < 4; ++j)
            lB[(c4 + j) * 64 + k] = (_Float16)Weff[k * DD + col0 + c4 + j];
    }
    __syncthreads();

    const int arow  = row0 + (lane & 15);
    const int koff  = (lane >> 4) << 4;
    const int bcolL = ((wid & 3) << 4) + (lane & 15);
    const int bkoff = (lane >> 4) << 4;
    const float dv  = dist[arow];

    v8f acc = {};
#pragma unroll
    for (int k0 = 0; k0 < RPAD; k0 += 32) {
        v16h av;
#pragma unroll
        for (int v = 0; v < 8; ++v) {
            const int kk = k0 + koff + (v << 1);
            av[2 * v]     = (_Float16)rbf_val(dv, kk);
            av[2 * v + 1] = (_Float16)rbf_val(dv, kk + 1);
        }
        const v16h bv = *(const v16h*)&lB[bcolL * 64 + k0 + bkoff];
        acc = __builtin_amdgcn_wmma_f32_16x16x32_f16(
            false, av, false, bv, (short)0, acc, false, false);
    }

    const int crow0 = row0 + ((lane >> 4) << 3);
    const int bcol  = col0 + bcolL;
    const float bb  = be[bcol];
#pragma unroll
    for (int r = 0; r < 8; ++r)
        E[(long)(crow0 + r) * DD + bcol] = acc[r] + bb;
}

// ---------------------------------------------------------------------------
// Stage A: grid = gelu(encoding @ W_up + b).reshape(G,DIN,NG).T -> [G,NG,DIN]
// ---------------------------------------------------------------------------
__global__ void upscale_kernel(const float* __restrict__ enc,
                               const float* __restrict__ Wup,
                               const float* __restrict__ bup,
                               float* __restrict__ grid) {
    const int t = blockIdx.x * 256 + threadIdx.x;
    if (t >= GG * NGP * DD) return;
    const int g = t / (NGP * DD);
    const int r = t - g * NGP * DD;
    const int n = r / DD;
    const int d = r - n * DD;
    const int col = d * NGP + n;
    float s = bup[col];
    const float* eg = enc + g * DD;
    for (int i = 0; i < DD; ++i) s += eg[i] * Wup[i * (DD * NGP) + col];
    grid[(g * NGP + n) * DD + d] = gelu_f(s);
}

// ---------------------------------------------------------------------------
// Stage B: knn_interpolate (k=3) from 27 gridpoints -> x[G,M,D], then gelu.
// ---------------------------------------------------------------------------
__global__ void interp_kernel(const float* __restrict__ pos,
                              const float* __restrict__ grid,
                              float* __restrict__ x) {
    const int t = blockIdx.x * 256 + threadIdx.x;
    if (t >= NODES * DD) return;
    const int node = t >> 7, d = t & 127;
    const int g = node >> 9;
    const float px = pos[node * 3], py = pos[node * 3 + 1], pz = pos[node * 3 + 2];

    float d2a[NGP];
#pragma unroll
    for (int n = 0; n < NGP; ++n) {
        const float gx = (float)(n / 9) - 1.0f;
        const float gy = (float)((n / 3) % 3) - 1.0f;
        const float gz = (float)(n % 3) - 1.0f;
        const float dx = px - gx, dy = py - gy, dz = pz - gz;
        d2a[n] = dx * dx + dy * dy + dz * dz;
    }
    int bi[3];
    float wsum = 0.0f, acc = 0.0f;
    for (int p = 0; p < 3; ++p) {
        float bv = 3.4e38f; int bn = 0;
        for (int n = 0; n < NGP; ++n) {
            bool taken = false;
            for (int q = 0; q < p; ++q) taken = taken || (bi[q] == n);
            if (!taken && d2a[n] < bv) { bv = d2a[n]; bn = n; }
        }
        bi[p] = bn;
        const float w = 1.0f / fmaxf(bv, 1e-16f);
        wsum += w;
        acc += w * grid[(g * NGP + bn) * DD + d];
    }
    x[node * DD + d] = gelu_f(acc / wsum);
}

// ---------------------------------------------------------------------------
// Top-K=32 neighbor search (self excluded via +1e9). One wave per node:
// each lane caches 16 candidate d2 in regs, 32 rounds of butterfly argmin.
// ---------------------------------------------------------------------------
__global__ void knn_kernel(const float* __restrict__ pos,
                           int* __restrict__ nbr,
                           float* __restrict__ dist) {
    const int wid  = threadIdx.x >> 5;
    const int lane = threadIdx.x & 31;
    const int node = blockIdx.x * 8 + wid;
    const int g = node >> 9, mm = node & 511;
    const float* pg = pos + g * MM * 3;
    const float px = pg[mm * 3], py = pg[mm * 3 + 1], pz = pg[mm * 3 + 2];

    float cd[16];
#pragma unroll
    for (int tc = 0; tc < 16; ++tc) {
        const int j = lane + 32 * tc;
        const float dx = pg[j * 3] - px;
        const float dy = pg[j * 3 + 1] - py;
        const float dz = pg[j * 3 + 2] - pz;
        float d2 = dx * dx + dy * dy + dz * dz;
        if (j == mm) d2 += 1e9f;
        cd[tc] = d2;
    }

    unsigned used = 0;
    for (int it = 0; it < KK; ++it) {
        float bv = 3.4e38f; int bj = 1 << 30;
#pragma unroll
        for (int tc = 0; tc < 16; ++tc) {
            if (!(used & (1u << tc))) {
                const int j = lane + 32 * tc;
                if (cd[tc] < bv) { bv = cd[tc]; bj = j; }
            }
        }
        for (int off = 16; off; off >>= 1) {
            const float ov = __shfl_xor(bv, off, 32);
            const int   oj = __shfl_xor(bj, off, 32);
            if (ov < bv || (ov == bv && oj < bj)) { bv = ov; bj = oj; }
        }
        if ((bj & 31) == lane) used |= 1u << (bj >> 5);
        if (lane == 0) {
            nbr[node * KK + it]  = bj;
            dist[node * KK + it] = sqrtf(fmaxf(bv, 1e-12f));
        }
    }
}

// ---------------------------------------------------------------------------
// TransformerConv attention: one wave per node, lane = neighbor (K==wave32).
// Phase 1: per-head masked softmax over 32 lanes (shfl reductions).
// Phase 2: roles flip -> lane = feature lane; attention weights via LDS.
// global_prefetch of next gathered v-row hides gather latency.
// ---------------------------------------------------------------------------
__global__ void attn_kernel(const float* __restrict__ qf,
                            const float* __restrict__ kf,
                            const float* __restrict__ vf,
                            const float* __restrict__ skf,
                            const float* __restrict__ e,
                            const int*   __restrict__ nbr,
                            const float* __restrict__ dist,
                            float* __restrict__ mout) {
    __shared__ float sh_a[8][HH][KK];
    __shared__ int   sh_n[8][KK];
    const int wid  = threadIdx.x >> 5;
    const int lane = threadIdx.x & 31;
    const int node = blockIdx.x * 8 + wid;
    const int g = node >> 9;

    const int   nidx = nbr[node * KK + lane];
    const float dk   = dist[node * KK + lane];
    const bool  msk  = (dk < CUT);
    const float* qrow = qf + node * DD;
    const float* krow = kf + (g * MM + nidx) * DD;
    const float* erow = e + (long)(node * KK + lane) * DD;

#pragma unroll
    for (int h = 0; h < HH; ++h) {
        float s = 0.0f;
        for (int d = 0; d < DH; ++d)
            s += qrow[h * DH + d] * (krow[h * DH + d] + erow[h * DH + d]);
        float sc = msk ? s * SCALE : -1e30f;
        float mx = sc;
        for (int off = 16; off; off >>= 1) mx = fmaxf(mx, __shfl_xor(mx, off, 32));
        const float ex = expf(sc - mx);
        float sm = ex;
        for (int off = 16; off; off >>= 1) sm += __shfl_xor(sm, off, 32);
        sh_a[wid][h][lane] = msk ? ex / sm : 0.0f;
    }
    sh_n[wid][lane] = nidx;
    __syncthreads();

    float acc[4] = {0.0f, 0.0f, 0.0f, 0.0f};
    for (int k = 0; k < KK; ++k) {
        const int nk = sh_n[wid][k];
        const float* vrow = vf + (g * MM + nk) * DD;
        if (k + 1 < KK)   // prefetch next gathered row (global_prefetch_b8)
            __builtin_prefetch(vf + (g * MM + sh_n[wid][k + 1]) * DD + lane, 0, 0);
        const float* er = e + (long)(node * KK + k) * DD;
#pragma unroll
        for (int c = 0; c < 4; ++c) {             // feature f = lane + 32c, head c
            const int f = lane + 32 * c;
            acc[c] += sh_a[wid][c][k] * (vrow[f] + er[f]);
        }
    }
#pragma unroll
    for (int c = 0; c < 4; ++c) {
        const int f = lane + 32 * c;
        mout[node * DD + f] = acc[c] + skf[node * DD + f];
    }
}

// ---------------------------------------------------------------------------
// Weff = Wrad[l] (50x128) @ We[l] (128x128), zero-padded to 64 rows.
// ---------------------------------------------------------------------------
__global__ void weff_kernel(const float* __restrict__ Wrad,
                            const float* __restrict__ We,
                            float* __restrict__ Weff) {
    const int t = blockIdx.x * 256 + threadIdx.x;
    if (t >= RPAD * DD) return;
    const int r = t >> 7, c = t & 127;
    float s = 0.0f;
    if (r < RR)
        for (int i = 0; i < DD; ++i) s += Wrad[r * DD + i] * We[i * DD + c];
    Weff[t] = s;
}

// ---------------------------------------------------------------------------
// Residual + position update: x+=h; pos+=x[:3]; x[-3:]=pos. One thread/node.
// ---------------------------------------------------------------------------
__global__ void update_kernel(float* __restrict__ x,
                              const float* __restrict__ h,
                              float* __restrict__ pos) {
    const int node = blockIdx.x * 256 + threadIdx.x;
    if (node >= NODES) return;
    float* xr = x + node * DD;
    const float* hr = h + node * DD;
    const float np0 = pos[node * 3]     + xr[0] + hr[0];
    const float np1 = pos[node * 3 + 1] + xr[1] + hr[1];
    const float np2 = pos[node * 3 + 2] + xr[2] + hr[2];
    for (int d = 0; d < DD; ++d) xr[d] = xr[d] + hr[d];
    xr[DD - 3] = np0; xr[DD - 2] = np1; xr[DD - 1] = np2;
    pos[node * 3] = np0; pos[node * 3 + 1] = np1; pos[node * 3 + 2] = np2;
}

__global__ void poscopy_kernel(const float* __restrict__ pos,
                               float* __restrict__ out) {
    const int t = blockIdx.x * 256 + threadIdx.x;
    if (t >= NODES * 3) return;
    out[(t / 3) * (3 + OUTN) + (t % 3)] = pos[t];
}

// ---------------------------------------------------------------------------
// Host-side orchestration
// ---------------------------------------------------------------------------
extern "C" void kernel_launch(void* const* d_in, const int* in_sizes, int n_in,
                              void* d_out, int out_size, void* d_ws, size_t ws_size,
                              hipStream_t stream) {
    (void)in_sizes; (void)n_in; (void)out_size; (void)ws_size;

    const float* enc    = (const float*)d_in[0];
    const float* pos_in = (const float*)d_in[1];
    const float* Wup    = (const float*)d_in[3];
    const float* bup    = (const float*)d_in[4];
    const float* Wnm    = (const float*)d_in[5];
    const float* Wmn    = (const float*)d_in[6];
    const float* Wrad   = (const float*)d_in[7];
    const float* We     = (const float*)d_in[8];
    const float* be     = (const float*)d_in[9];
    const float* Wq     = (const float*)d_in[10];
    const float* bq     = (const float*)d_in[11];
    const float* Wk     = (const float*)d_in[12];
    const float* bk     = (const float*)d_in[13];
    const float* Wv     = (const float*)d_in[14];
    const float* bv     = (const float*)d_in[15];
    const float* Wskip  = (const float*)d_in[16];
    const float* bskip  = (const float*)d_in[17];
    const float* Wfc    = (const float*)d_in[18];
    const float* bfc    = (const float*)d_in[19];
    const float* Wfco   = (const float*)d_in[20];
    const float* bfco   = (const float*)d_in[21];
    const float* Wout   = (const float*)d_in[22];
    const float* bout   = (const float*)d_in[23];
    float* out = (float*)d_out;

    // workspace carve (~175 MB of f32; dominated by e[E,128])
    float* w = (float*)d_ws;
    size_t o = 0;
    float* gridf = w + o; o += (size_t)GG * NGP * DD;
    float* x     = w + o; o += (size_t)NODES * DD;
    float* posb  = w + o; o += (size_t)NODES * 3;
    float* dist  = w + o; o += (size_t)EDGES;
    int*   nbr   = (int*)(w + o); o += (size_t)EDGES;
    float* mbuf  = w + o; o += (size_t)NODES * DD;
    float* qf    = w + o; o += (size_t)NODES * DD;
    float* kf    = w + o; o += (size_t)NODES * DD;
    float* vf    = w + o; o += (size_t)NODES * DD;
    float* skf   = w + o; o += (size_t)NODES * DD;
    float* moutb = w + o; o += (size_t)NODES * DD;
    float* ha    = w + o; o += (size_t)NODES * DD;
    float* hb    = w + o; o += (size_t)NODES * DD;
    float* weff  = w + o; o += (size_t)RPAD * DD;
    float* ebuf  = w + o; o += (size_t)EDGES * DD;

    const int DD2 = DD * DD;
    const dim3 blk(256);
    const dim3 gemm_grid(NODES / 32, DD / 64);      // [8192,128] GEMMs
    const dim3 edge_grid(EDGES / 32, DD / 64);      // [262144,128] edge GEMM
    const dim3 out_grid(NODES / 32, (OUTN + 63) / 64);

    // working copy of positions (inputs must not be mutated)
    hipMemcpyAsync(posb, pos_in, (size_t)NODES * 3 * sizeof(float),
                   hipMemcpyDeviceToDevice, stream);

    upscale_kernel<<<(GG * NGP * DD + 255) / 256, blk, 0, stream>>>(enc, Wup, bup, gridf);
    interp_kernel<<<(NODES * DD + 255) / 256, blk, 0, stream>>>(posb, gridf, x);

    for (int l = 0; l < LL; ++l) {
        knn_kernel<<<NODES / 8, blk, 0, stream>>>(posb, nbr, dist);

        wmma_gemm_kernel<<<gemm_grid, blk, 0, stream>>>(x, Wnm + l * DD2, nullptr, mbuf,
                                                        DD, DD, DD, 0);
        wmma_gemm_kernel<<<gemm_grid, blk, 0, stream>>>(mbuf, Wq + l * DD2, bq + l * DD, qf,
                                                        DD, DD, DD, 0);
        wmma_gemm_kernel<<<gemm_grid, blk, 0, stream>>>(mbuf, Wk + l * DD2, bk + l * DD, kf,
                                                        DD, DD, DD, 0);
        wmma_gemm_kernel<<<gemm_grid, blk, 0, stream>>>(mbuf, Wv + l * DD2, bv + l * DD, vf,
                                                        DD, DD, DD, 0);
        wmma_gemm_kernel<<<gemm_grid, blk, 0, stream>>>(mbuf, Wskip + l * DD2, bskip + l * DD, skf,
                                                        DD, DD, DD, 0);

        weff_kernel<<<(RPAD * DD + 255) / 256, blk, 0, stream>>>(Wrad + l * RR * DD,
                                                                 We + l * DD2, weff);
        edge_wmma_gemm_kernel<<<edge_grid, blk, 0, stream>>>(dist, weff, be + l * DD, ebuf);

        attn_kernel<<<NODES / 8, blk, 0, stream>>>(qf, kf, vf, skf, ebuf, nbr, dist, moutb);

        wmma_gemm_kernel<<<gemm_grid, blk, 0, stream>>>(moutb, Wmn + l * DD2, nullptr, ha,
                                                        DD, DD, DD, 0);
        wmma_gemm_kernel<<<gemm_grid, blk, 0, stream>>>(ha, Wfc + (l * 2 + 0) * DD2,
                                                        bfc + (l * 2 + 0) * DD, hb,
                                                        DD, DD, DD, 1);
        wmma_gemm_kernel<<<gemm_grid, blk, 0, stream>>>(hb, Wfc + (l * 2 + 1) * DD2,
                                                        bfc + (l * 2 + 1) * DD, ha,
                                                        DD, DD, DD, 1);
        wmma_gemm_kernel<<<gemm_grid, blk, 0, stream>>>(ha, Wfco + l * DD2, bfco + l * DD, hb,
                                                        DD, DD, DD, 0);

        update_kernel<<<(NODES + 255) / 256, blk, 0, stream>>>(x, hb, posb);
    }

    poscopy_kernel<<<(NODES * 3 + 255) / 256, blk, 0, stream>>>(posb, out);
    wmma_gemm_kernel<<<out_grid, blk, 0, stream>>>(x, Wout, bout, out + 3,
                                                   OUTN, OUTN, 3 + OUTN, 0);
}
